// Net_3092376453262
// MI455X (gfx1250) — matrix operand order repaired and checked by reading, not compiled, not run on hardware
//
#include <hip/hip_runtime.h>
#include <hip/hip_bf16.h>

// ---------------------------------------------------------------------------
// 3-layer GRU (B=64, S=512, H=512, IN=64) + FC(512->24), PyTorch semantics.
// f16 storage / f32 WMMA accumulate. Persistent kernel per layer:
// 32 WGs x 128 threads; each WG owns 16 hidden units (48 gate rows); each of
// the 4 waves owns a 16-row batch tile and 3 N-tiles (r,z,n gates).
// - W_hh slice: TDM -> LDS once, reused for all 512 timesteps.
// - h state:    TDM -> LDS every timestep, overlapped with the x-side GEMM,
//               so the recurrent GEMM is all-LDS (ds_load_b128 A and B frags).
// - per-timestep grid sync via L2 atomic barrier + s_sleep.
// Weights pre-converted to f16 row-major [3H][K] so WMMA B-fragments are
// contiguous 32B loads in the contiguous-K lane order used for A.
// ---------------------------------------------------------------------------

typedef _Float16     v8h   __attribute__((ext_vector_type(8)));
typedef _Float16     v16h  __attribute__((ext_vector_type(16)));
typedef float        v8f   __attribute__((ext_vector_type(8)));
typedef unsigned int u32x4 __attribute__((ext_vector_type(4)));
typedef int          i32x4 __attribute__((ext_vector_type(4)));
typedef int          i32x8 __attribute__((ext_vector_type(8)));

#define HID   512
#define BATCH 64
#define SEQ   512
#define NWG   32     // workgroups in recurrent kernel (each owns 16 units)
#define TOUT  24

#if defined(__has_builtin)
#  if __has_builtin(__builtin_amdgcn_tensor_load_to_lds)
#    define HAVE_TDM 1
#  endif
#endif
#ifndef HAVE_TDM
#  define HAVE_TDM 0
#endif

__device__ __forceinline__ v16h cat8(v8h lo, v8h hi) {
  return __builtin_shufflevector(lo, hi, 0,1,2,3,4,5,6,7,8,9,10,11,12,13,14,15);
}

// A-matrix fragment (16x32 f16, M x K), src row-major with leading dim `ld`.
// Layout: lanes 0-15: M=lane, K = k+{0..15};
//         lanes 16-31: M=lane-16, K = k+{16..31}.
__device__ __forceinline__ v16h load_a_frag(const _Float16* base, int row,
                                            int ld, int k, int lane) {
  const int r    = row + (lane & 15);
  const int koff = (lane & 16) ? 16 : 0;
  const _Float16* p = base + (size_t)r * ld + k + koff;
  v8h lo = *(const v8h*)p;
  v8h hi = *(const v8h*)(p + 8);
  return cat8(lo, hi);
}

// B-matrix fragment (32x16 f16, K x N) where B[k][n] = W[grow+n][k],
// W row-major [G][ld]. Layout used here (same as load_a_frag): lane n=lane%16 holds 16 contiguous K
// halves: lanes 0-15 -> K = k..k+15, lanes 16-31 -> K = k+16..k+31.
// => one contiguous 32B load per lane from W's row (ds_load/global_load b128).
__device__ __forceinline__ v16h load_b_frag(const _Float16* base, int grow,
                                            int ld, int k, int lane) {
  const int g    = grow + (lane & 15);
  const int koff = (lane & 16) ? 16 : 0;
  return *(const v16h*)(base + (size_t)g * ld + k + koff);
}

#define WMMA_F16(a, b, c) \
  __builtin_amdgcn_wmma_f32_16x16x32_f16(false, (a), false, (b), (short)0, (c), false, false)

__device__ __forceinline__ float sigm(float x) {
  return 1.0f / (1.0f + __expf(-x));
}

#if HAVE_TDM
// 1-D flat TDM copy, global -> LDS. data_size = 8B; nbytes multiple of 8,
// nbytes/8 must fit tile_dim0's 16 bits (<= 512KB). D# per ISA 08_async_tensor:
// group0: count=1 | lds_addr[63:32] | global_addr[120:64] | type=2 [127:126]
// group1: data_size=3 (8B) [17:16]; tensor_dim0 [79:48]; tensor_dim1=1 [111:80];
//         tile_dim0 [127:112]; tile_dim1=1 [143:128]; dim0_stride [207:160].
__device__ __forceinline__ void tdm_load_1d(void* lds_dst, const void* gsrc,
                                            unsigned nbytes) {
  const unsigned lds_addr = (unsigned)(uintptr_t)lds_dst; // low 32b = LDS offset
  const unsigned long long ga = (unsigned long long)(uintptr_t)gsrc;
  const unsigned n8 = nbytes >> 3;
  u32x4 g0;
  g0[0] = 1u;                                            // count=1 (valid, user)
  g0[1] = lds_addr;
  g0[2] = (unsigned)(ga & 0xFFFFFFFFu);
  g0[3] = (unsigned)((ga >> 32) & 0x01FFFFFFu) | (2u << 30); // type=2 ("image")
  i32x8 g1;
  g1[0] = (int)(3u << 16);                               // data_size = 8 bytes
  g1[1] = (int)((n8 & 0xFFFFu) << 16);                   // tensor_dim0[15:0]
  g1[2] = (int)(((n8 >> 16) & 0xFFFFu) | (1u << 16));    // tensor_dim0[31:16], tensor_dim1=1
  g1[3] = (int)((n8 & 0xFFFFu) << 16);                   // tile_dim0
  g1[4] = 1;                                             // tile_dim1 = 1
  g1[5] = (int)n8;                                       // tensor_dim0_stride
  g1[6] = 0;
  g1[7] = 0;
  i32x4 gz = {0, 0, 0, 0};
#if __clang_major__ >= 23
  i32x8 gz8 = {0, 0, 0, 0, 0, 0, 0, 0};
  __builtin_amdgcn_tensor_load_to_lds(g0, g1, gz, gz, gz8, 0);
#else
  __builtin_amdgcn_tensor_load_to_lds(g0, g1, gz, gz, 0);
#endif
}
#endif

// ---------------------------------------------------------------------------
// Persistent GRU layer kernel. xT: [S][B][IN_DIM] f16 (time-major).
// wih: [3H][IN_DIM] f16 (L2-resident, streamed), whh: [3H][H] f16 (TDM -> LDS).
// yT: [S][B][H] f16 output. hbuf0/hbuf1: [B][H] f16 ping-pong.
// ---------------------------------------------------------------------------
template <int IN_DIM>
__global__ __launch_bounds__(128, 1)
void gru_layer_kernel(const _Float16* __restrict__ xT,
                      const _Float16* __restrict__ wih,
                      const _Float16* __restrict__ whh,
                      const float* __restrict__ bih,
                      const float* __restrict__ bhh,
                      _Float16* __restrict__ hbuf0,
                      _Float16* __restrict__ hbuf1,
                      _Float16* __restrict__ yT,
                      unsigned* __restrict__ bar_cnt) {
  __shared__ _Float16 s_whh[48 * HID];       // this WG's W_hh slice (48 KB)
  __shared__ _Float16 s_h[BATCH * HID];      // h state tile, re-staged per step (64 KB)
  __shared__ float    s_bih[48];
  __shared__ float    s_bhh[48];

  const int tid   = threadIdx.x;
  const int wg    = blockIdx.x;       // 0..31
  const int unit0 = wg * 16;          // this WG's hidden-unit base
  const int wave  = tid >> 5;         // 0..3 -> batch tile base
  const int lane  = tid & 31;
  const int m0    = wave * 16;

  // Bias slices (local index i = gate*16 + local_unit).
  for (int i = tid; i < 48; i += 128) {
    const int g = (i >> 4) * HID + unit0 + (i & 15);
    s_bih[i] = bih[g];
    s_bhh[i] = bhh[g];
  }

  // Stage W_hh slice into LDS once; reused for all 512 timesteps.
  // Each gate's 16 rows (g = gate*H + unit0 ..) are contiguous 16KB in memory.
#if HAVE_TDM
  if (tid == 0) {
    for (int gate = 0; gate < 3; ++gate)
      tdm_load_1d(&s_whh[gate * 16 * HID],
                  whh + ((size_t)(gate * HID + unit0)) * HID,
                  16 * HID * 2);
    __builtin_amdgcn_s_wait_tensorcnt(0);
  }
#else
  for (int v = tid; v < 48 * HID / 8; v += 128) {
    const int rrow = v / (HID / 8);
    const int c8   = (v % (HID / 8)) * 8;
    const int g    = (rrow >> 4) * HID + unit0 + (rrow & 15);
    *(v8h*)&s_whh[rrow * HID + c8] = *(const v8h*)&whh[(size_t)g * HID + c8];
  }
#endif
  __syncthreads();

  for (int t = 0; t < SEQ; ++t) {
    const _Float16* hcur  = (t & 1) ? hbuf1 : hbuf0;
    _Float16*       hnext = (t & 1) ? hbuf0 : hbuf1;

    // Kick off the h -> LDS DMA for this step (64 KB, one TDM op), then
    // overlap it with the input-side GEMM which doesn't depend on h.
#if HAVE_TDM
    if (tid == 0) tdm_load_1d(s_h, hcur, BATCH * HID * 2);
#else
    for (int v = tid; v < BATCH * HID / 8; v += 128)
      *(v8h*)&s_h[v * 8] = *(const v8h*)&hcur[v * 8];
#endif

    v8f ax[3]; // input-side accumulators, nt = gate (r,z,n)
    v8f ah[3]; // hidden-side accumulators
#pragma unroll
    for (int i = 0; i < 3; ++i) { ax[i] = {}; ah[i] = {}; }

    // ---- input-side GEMM: [64 x IN_DIM] x [IN_DIM x 48] (W_ih from L2) ----
    const _Float16* xrow = xT + (size_t)t * (BATCH * IN_DIM);
#pragma unroll 2
    for (int k = 0; k < IN_DIM; k += 32) {
      const v16h a = load_a_frag(xrow, m0, IN_DIM, k, lane);
#pragma unroll
      for (int nt = 0; nt < 3; ++nt) {
        const v16h b = load_b_frag(wih, nt * HID + unit0, IN_DIM, k, lane);
        ax[nt] = WMMA_F16(a, b, ax[nt]);
      }
    }

    // Publish the staged h tile to the whole workgroup.
#if HAVE_TDM
    if (tid == 0) __builtin_amdgcn_s_wait_tensorcnt(0);
#endif
    __syncthreads();

    // ---- hidden-side GEMM: [64 x 512] x [512 x 48], all-LDS ---------------
#pragma unroll 4
    for (int k = 0; k < HID; k += 32) {
      const v16h a = load_a_frag(s_h, m0, HID, k, lane);
#pragma unroll
      for (int nt = 0; nt < 3; ++nt) {
        const v16h b = load_b_frag(s_whh, nt * 16, HID, k, lane);
        ah[nt] = WMMA_F16(a, b, ah[nt]);
      }
    }

    // ---- gate combine + state update --------------------------------------
    // C/D layout: VGPR r holds M = r (lanes 0-15) / r+8 (lanes 16-31), N = lane%16.
    const int n  = lane & 15;
    const int mh = (lane & 16) ? 8 : 0;
    const int   unit = unit0 + n;
    const float br   = s_bih[n] + s_bhh[n];
    const float bz   = s_bih[16 + n] + s_bhh[16 + n];
    const float bin_ = s_bih[32 + n];
    const float bhn  = s_bhh[32 + n];
#pragma unroll
    for (int r = 0; r < 8; ++r) {
      const int   mrow = m0 + mh + r;
      const float rv = sigm(ax[0][r] + ah[0][r] + br);
      const float zv = sigm(ax[1][r] + ah[1][r] + bz);
      const float nv = tanhf(ax[2][r] + bin_ + rv * (ah[2][r] + bhn));
      const float hp = (float)s_h[(size_t)mrow * HID + unit];   // from LDS tile
      const float hv = (1.0f - zv) * nv + zv * hp;
      const _Float16 h16 = (_Float16)hv;
      hnext[(size_t)mrow * HID + unit] = h16;
      yT[((size_t)t * BATCH + mrow) * HID + unit] = h16;
    }

    // Warm WGP$ with next timestep's input row while we wait at the barrier.
    if (t + 1 < SEQ) {
      const _Float16* nx = xT + (size_t)(t + 1) * (BATCH * IN_DIM);
      __builtin_prefetch(nx + (size_t)tid * (BATCH * IN_DIM / 128), 0, 1);
    }

    // ---- grid-wide barrier (monotonic counter in L2) ----------------------
    __threadfence();            // release our h' slice to agent scope
    __syncthreads();
    if (tid == 0) {
      atomicAdd(bar_cnt, 1u);
      const unsigned target = (unsigned)(t + 1) * (unsigned)NWG;
      while (__hip_atomic_load(bar_cnt, __ATOMIC_ACQUIRE,
                               __HIP_MEMORY_SCOPE_AGENT) < target) {
        __builtin_amdgcn_s_sleep(2);
      }
    }
    __syncthreads();
    __threadfence();            // invalidate L0: h ping-pong buffer re-read
  }
}

// ---------------------------------------------------------------------------
// Prep / epilogue kernels
// ---------------------------------------------------------------------------
__global__ void cvt_f32_to_f16(const float* __restrict__ src,
                               _Float16* __restrict__ dst, int n) {
  const int i = blockIdx.x * blockDim.x + threadIdx.x;
  if (i < n) dst[i] = (_Float16)src[i];
}

__global__ void transpose_x_kernel(const float* __restrict__ x,
                                   _Float16* __restrict__ xT) {
  // x: [B][S][64] -> xT: [S][B][64]
  const int i = blockIdx.x * blockDim.x + threadIdx.x;
  if (i < BATCH * SEQ * 64) {
    const int in_ = i & 63;
    const int s   = (i >> 6) & (SEQ - 1);
    const int b   = i / (64 * SEQ);
    xT[((size_t)s * BATCH + b) * 64 + in_] = (_Float16)x[i];
  }
}

__global__ void init_layer_kernel(_Float16* __restrict__ h0,
                                  unsigned* __restrict__ bar) {
  const int i = blockIdx.x * blockDim.x + threadIdx.x;
  if (i < BATCH * HID) h0[i] = (_Float16)0.0f;
  if (i == 0) *bar = 0u;
}

__global__ void fc_kernel(const _Float16* __restrict__ yT, // [S][B][H]
                          const float* __restrict__ fcw,   // [24][512]
                          const float* __restrict__ fcb,
                          float* __restrict__ out) {
  const int i = blockIdx.x * blockDim.x + threadIdx.x;
  if (i >= BATCH * TOUT) return;
  const int o = i % TOUT;
  const int b = i / TOUT;
  const _Float16* hrow = yT + ((size_t)(SEQ - 1) * BATCH + b) * HID;
  float acc = fcb[o];
  for (int k = 0; k < HID; ++k)
    acc += (float)hrow[k] * fcw[(size_t)o * HID + k];
  out[b * TOUT + o] = acc;
}

// ---------------------------------------------------------------------------
extern "C" void kernel_launch(void* const* d_in, const int* in_sizes, int n_in,
                              void* d_out, int out_size, void* d_ws, size_t ws_size,
                              hipStream_t stream) {
  (void)in_sizes; (void)n_in; (void)out_size; (void)ws_size;
  const float* x      = (const float*)d_in[0];
  const float* wih[3] = {(const float*)d_in[1], (const float*)d_in[5], (const float*)d_in[9]};
  const float* whh[3] = {(const float*)d_in[2], (const float*)d_in[6], (const float*)d_in[10]};
  const float* bih[3] = {(const float*)d_in[3], (const float*)d_in[7], (const float*)d_in[11]};
  const float* bhh[3] = {(const float*)d_in[4], (const float*)d_in[8], (const float*)d_in[12]};
  const float* fcw    = (const float*)d_in[13];
  const float* fcb    = (const float*)d_in[14];
  float* out          = (float*)d_out;

  // ---- workspace layout (~79.5 MB total, 256B aligned chunks) ----
  char* ws = (char*)d_ws;
  size_t off = 0;
  auto carve = [&](size_t bytes) -> char* {
    char* p = ws + off;
    off = (off + bytes + 255) & ~(size_t)255;
    return p;
  };
  _Float16* xT   = (_Float16*)carve((size_t)SEQ * BATCH * 64 * 2);    //  4 MB
  _Float16* actA = (_Float16*)carve((size_t)SEQ * BATCH * HID * 2);   // 32 MB
  _Float16* actB = (_Float16*)carve((size_t)SEQ * BATCH * HID * 2);   // 32 MB
  _Float16* wih16[3], *whh16[3];
  wih16[0] = (_Float16*)carve((size_t)3 * HID * 64 * 2);
  wih16[1] = (_Float16*)carve((size_t)3 * HID * HID * 2);
  wih16[2] = (_Float16*)carve((size_t)3 * HID * HID * 2);
  for (int l = 0; l < 3; ++l) whh16[l] = (_Float16*)carve((size_t)3 * HID * HID * 2);
  _Float16* h0  = (_Float16*)carve((size_t)BATCH * HID * 2);
  _Float16* h1  = (_Float16*)carve((size_t)BATCH * HID * 2);
  unsigned* bar = (unsigned*)carve(256);

  // ---- precision conversion + time-major transpose ----
  {
    const int nx = BATCH * SEQ * 64;
    transpose_x_kernel<<<(nx + 255) / 256, 256, 0, stream>>>(x, xT);
    const int nwih0 = 3 * HID * 64;
    const int nbig  = 3 * HID * HID;
    cvt_f32_to_f16<<<(nwih0 + 255) / 256, 256, 0, stream>>>(wih[0], wih16[0], nwih0);
    cvt_f32_to_f16<<<(nbig + 255) / 256, 256, 0, stream>>>(wih[1], wih16[1], nbig);
    cvt_f32_to_f16<<<(nbig + 255) / 256, 256, 0, stream>>>(wih[2], wih16[2], nbig);
    for (int l = 0; l < 3; ++l)
      cvt_f32_to_f16<<<(nbig + 255) / 256, 256, 0, stream>>>(whh[l], whh16[l], nbig);
  }

  const int ninit = BATCH * HID;
  const dim3 gridGRU(NWG), blockGRU(128);

  // ---- layer 0: xT -> actA ----
  init_layer_kernel<<<(ninit + 255) / 256, 256, 0, stream>>>(h0, bar);
  gru_layer_kernel<64><<<gridGRU, blockGRU, 0, stream>>>(
      xT, wih16[0], whh16[0], bih[0], bhh[0], h0, h1, actA, bar);
  // ---- layer 1: actA -> actB ----
  init_layer_kernel<<<(ninit + 255) / 256, 256, 0, stream>>>(h0, bar);
  gru_layer_kernel<512><<<gridGRU, blockGRU, 0, stream>>>(
      actA, wih16[1], whh16[1], bih[1], bhh[1], h0, h1, actB, bar);
  // ---- layer 2: actB -> actA ----
  init_layer_kernel<<<(ninit + 255) / 256, 256, 0, stream>>>(h0, bar);
  gru_layer_kernel<512><<<gridGRU, blockGRU, 0, stream>>>(
      actB, wih16[2], whh16[2], bih[2], bhh[2], h0, h1, actA, bar);

  // ---- FC epilogue on h[:, -1, :] ----
  fc_kernel<<<(BATCH * TOUT + 255) / 256, 256, 0, stream>>>(actA, fcw, fcb, out);
}
